// RealSpaceBackflow_9594956939552
// MI455X (gfx1250) — matrix operand order repaired and checked by reading, not compiled, not run on hardware
//
#include <hip/hip_runtime.h>
#include <stdint.h>

// ---------------------------------------------------------------------------
// RealSpaceBackflow for MI455X (gfx1250, wave32, WMMA).
//
// Roofline: first-layer GEMMs = ~84 GFLOP vs 335 MB unique bytes (14.4 us at
// 23.3 TB/s). f16 WMMA w/ f32 accumulate balances compute vs memory.
// This revision: (a) software-pipelined inner loop (B-fragment register
// double-buffer + preloaded A fragments) to kill the per-WMMA s_wait_dscnt
// stalls, (b) double-buffered TDM weight DMA with counted s_wait_tensorcnt,
// (c) non-temporal loads for the stream-once message tensor.
// ---------------------------------------------------------------------------

typedef __attribute__((ext_vector_type(16))) _Float16 v16h;
typedef __attribute__((ext_vector_type(8)))  _Float16 v8h;
typedef __attribute__((ext_vector_type(4)))  _Float16 v4h;
typedef __attribute__((ext_vector_type(8)))  float    v8f;
typedef __attribute__((ext_vector_type(4)))  float    v4f;
typedef __attribute__((ext_vector_type(4)))  unsigned v4u;
typedef __attribute__((ext_vector_type(8)))  unsigned v8u;

#define N_PAIRS  16384        // B * N_ELEC

// -------------------------------------------------------------------
// Kernel P: transpose + f32->f16 convert the two 256x256 weight mats.
// Output layout wt[n][k] so WMMA B-fragments are contiguous in K.
// -------------------------------------------------------------------
__global__ __launch_bounds__(256) void prep_weights(
    const float* __restrict__ w1el, const float* __restrict__ w1nuc,
    _Float16* __restrict__ wtel, _Float16* __restrict__ wtnuc)
{
    int i = blockIdx.x * 256 + threadIdx.x;          // 0 .. 65535
    if (i < 256 * 256) {
        int n = i >> 8, k = i & 255;
        wtel [n * 256 + k] = (_Float16)w1el [k * 256 + n];
        wtnuc[n * 256 + k] = (_Float16)w1nuc[k * 256 + n];
    }
}

// -------------------------------------------------------------------
// TDM: DMA a 256-row x 128-half (64KB) f16 tile, row stride 256 halves
// in memory, packed (stride 128) into LDS. Issued by one wave; tracked
// by TENSORcnt.
// -------------------------------------------------------------------
__device__ __forceinline__ void tdm_load_w_chunk(const _Float16* gsrc,
                                                 unsigned lds_off)
{
    unsigned long long ga = (unsigned long long)(uintptr_t)gsrc;
    v4u g0;
    g0.x = 1u;                                        // count=1, user mode
    g0.y = lds_off;                                   // LDS byte address
    g0.z = (unsigned)(ga & 0xFFFFFFFFull);            // global_addr[31:0]
    g0.w = (unsigned)((ga >> 32) & 0x1FFFFFFull) | (2u << 30); // hi | type=2
    v8u g1;
    g1[0] = 1u << 16;            // workgroup_mask=0, data_size=1 (2 bytes)
    g1[1] = (256u & 0xFFFFu) << 16;                   // tensor_dim0 lo16
    g1[2] = 0u | (256u << 16);   // tensor_dim0 hi16=0, tensor_dim1 lo16
    g1[3] = 128u << 16;          // tensor_dim1 hi16=0, tile_dim0 = 128
    g1[4] = 256u;                // tile_dim1 = 256, tile_dim2 = 0
    g1[5] = 256u;                // tensor_dim0_stride lo32 = 256
    g1[6] = 0u;
    g1[7] = 0u;
    v4u g2; g2.x = 1u; g2.y = 1u; g2.z = 0u; g2.w = 0u;       // dim2=1, dim3=1
    v4u g3; g3.x = 0u; g3.y = 1u << 16; g3.z = 0u; g3.w = 0u; // dim4=1
    asm volatile("tensor_load_to_lds %0, %1, %2, %3"
                 :: "s"(g0), "s"(g1), "s"(g2), "s"(g3)
                 : "memory");
}

// Fragment loaders. A: lane holds M=l15, K = base + lh*16 + (0..15).
// B: lane holds N=l15, K = base + lh*16 + (0..15).
__device__ __forceinline__ v16h load_a_frag(const _Float16* p) {
    v8h x0 = *(const v8h*)p;
    v8h x1 = *(const v8h*)(p + 8);
    return __builtin_shufflevector(x0, x1, 0,1,2,3,4,5,6,7,8,9,10,11,12,13,14,15);
}
__device__ __forceinline__ v16h load_b_frag(const _Float16* p) {
    v8h x0 = *(const v8h*)p;
    v8h x1 = *(const v8h*)(p + 8);
    return __builtin_shufflevector(x0, x1, 0,1,2,3,4,5,6,7,8,9,10,11,12,13,14,15);
}

// -------------------------------------------------------------------
// Kernel G: fused first+second MLP layer.
//   RPG = rows per (b,e) pair (31 el / 8 nuc), PPW = pairs per WG.
//   M = 256 rows per WG (RPG*PPW real, rest zero-padded), N = 256,
//   K = 256 in two 128-wide phases (phase0 = emb, phase1 = messages).
//   16 waves: wave = M-tile, each owns all 16 N-tiles (128 acc VGPRs).
// -------------------------------------------------------------------
template<int RPG, int PPW>
__global__ __launch_bounds__(512, 1) void mlp_gemm(
    const float*    __restrict__ emb,     // [N_PAIRS][128] f32
    const float*    __restrict__ msgs,    // [N_PAIRS][RPG][128] f32
    const _Float16* __restrict__ wt,      // [256][256] f16 transposed
    const float*    __restrict__ b1,      // [256]
    const float*    __restrict__ w2,      // [256]
    const float*    __restrict__ b2,      // [1]
    float*          __restrict__ fout,    // [N_PAIRS][39]
    int jbase)                            // 0 for el, 31 for nuc
{
    extern __shared__ __align__(16) char smem[];
    _Float16* Alds = (_Float16*)smem;              // 256 x 128 f16 = 64 KB
    _Float16* Wlds0 = Alds + 256 * 128;            // 64 KB (K chunk 0)
    _Float16* Wlds1 = Wlds0 + 256 * 128;           // 64 KB (K chunk 1)

    const int tid  = threadIdx.x;
    const int lane = tid & 31;
    const int wave = tid >> 5;                     // 0..15 -> M-tile
    const int l15  = lane & 15;
    const int lh   = lane >> 4;
    const int pair0 = blockIdx.x * PPW;

    // Kick off BOTH weight-chunk DMAs up front; they run in the background
    // behind A staging / phase-0 compute. Counted TENSORcnt waits below.
    if (tid < 32) {
        tdm_load_w_chunk(wt,       (unsigned)(uintptr_t)Wlds0);
        tdm_load_w_chunk(wt + 128, (unsigned)(uintptr_t)Wlds1);
    }

    // Hint-prefetch this WG's message block (read once, streamed).
    {
        const char* mb = (const char*)(msgs + (size_t)pair0 * RPG * 128);
        const int lines = (RPG * PPW * 128 * 4) / 128;
        for (int it = tid; it < lines; it += 512)
            __builtin_prefetch(mb + (size_t)it * 128, 0, 1);
    }

    v8f acc[16];
#pragma unroll
    for (int i = 0; i < 16; ++i) {
        v8f z = {0.f, 0.f, 0.f, 0.f, 0.f, 0.f, 0.f, 0.f};
        acc[i] = z;
    }

#pragma unroll
    for (int kc = 0; kc < 2; ++kc) {
        // ---- stage A chunk: 256 rows x 128 cols, f32 -> f16 into LDS ----
        for (int it = tid; it < 256 * 32; it += 512) {
            int r  = it >> 5;             // row 0..255
            int c4 = it & 31;             // float4 index within row
            v4f val = {0.f, 0.f, 0.f, 0.f};
            if (r < RPG * PPW) {
                int p = r / RPG, j = r - p * RPG;
                if (kc == 0) {
                    val = *(const v4f*)(emb + ((size_t)(pair0 + p)) * 128 + c4 * 4);
                } else {
                    // stream-once data: non-temporal, keep L2 for weights
                    val = __builtin_nontemporal_load(
                        (const v4f*)(msgs +
                            (((size_t)(pair0 + p)) * RPG + j) * 128 + c4 * 4));
                }
            }
            v4h hv = {(_Float16)val.x, (_Float16)val.y,
                      (_Float16)val.z, (_Float16)val.w};
            *(v4h*)&Alds[r * 128 + c4 * 4] = hv;
        }

        // Phase 0 may leave the chunk-1 DMA in flight (tensorcnt <= 1).
        // Builtin requires a literal immediate -> branch on phase.
        if (tid < 32) {
            if (kc == 0) __builtin_amdgcn_s_wait_tensorcnt(1);
            else         __builtin_amdgcn_s_wait_tensorcnt(0);
        }
        __syncthreads();

        const _Float16* Wl = (kc == 0) ? Wlds0 : Wlds1;

        // ---- preload all 4 A fragments for this phase (32 VGPRs) ----
        v16h af[4];
#pragma unroll
        for (int ks = 0; ks < 4; ++ks)
            af[ks] = load_a_frag(&Alds[(wave * 16 + l15) * 128 + ks * 32 + lh * 16]);

        // ---- 64 WMMAs, software-pipelined B-fragment double buffer ----
        v16h bf = load_b_frag(&Wl[(0 * 16 + l15) * 128 + 0 * 32 + lh * 16]);
#pragma unroll
        for (int idx = 0; idx < 64; ++idx) {
            const int ks = idx >> 4, nt = idx & 15;
            v16h bn = bf;
            if (idx < 63) {
                const int ksn = (idx + 1) >> 4, ntn = (idx + 1) & 15;
                bn = load_b_frag(&Wl[(ntn * 16 + l15) * 128 + ksn * 32 + lh * 16]);
            }
            acc[nt] = __builtin_amdgcn_wmma_f32_16x16x32_f16(
                false, af[ks], false, bf, (short)0, acc[nt], false, false);
            bf = bn;
        }
        __syncthreads();
    }

    // ---- epilogue: bias + silu + second layer (dot with w2) ----
    // C layout: element (v, lane) -> M = v + 8*lh, N = l15 (per N-tile).
    float partial[8];
#pragma unroll
    for (int v = 0; v < 8; ++v) {
        float s = 0.f;
#pragma unroll
        for (int nt = 0; nt < 16; ++nt) {
            int n = nt * 16 + l15;
            float pre = acc[nt][v] + b1[n];
            float h = pre / (1.f + __expf(-pre));   // silu
            s += h * w2[n];
        }
        s += __shfl_xor(s, 1);
        s += __shfl_xor(s, 2);
        s += __shfl_xor(s, 4);
        s += __shfl_xor(s, 8);                      // reduce 16-lane half
        partial[v] = s;
    }
    if (l15 == 0) {
        float bias2 = b2[0];
#pragma unroll
        for (int v = 0; v < 8; ++v) {
            int r = wave * 16 + v + 8 * lh;
            if (r < RPG * PPW) {
                int p = r / RPG, j = r - p * RPG;
                fout[(size_t)(pair0 + p) * 39 + jbase + j] = partial[v] + bias2;
            }
        }
    }
}

// -------------------------------------------------------------------
// Kernel D: per-(b,e) epilogue — pairwise diffs, 1/(1+|d|^3) weights,
// sum over 39 rows, tanh decay product, final scale.
// -------------------------------------------------------------------
__global__ __launch_bounds__(256) void finalize(
    const float* __restrict__ rs,        // [512][32][3]
    const float* __restrict__ coords,    // [8][3]
    const float* __restrict__ chg,       // [8]
    const float* __restrict__ fws,       // [N_PAIRS][39]
    float*       __restrict__ out)       // [512][32][3]
{
    int pair = blockIdx.x * 256 + threadIdx.x;
    if (pair >= N_PAIRS) return;
    int b = pair >> 5, e = pair & 31;

    const float* rb = rs + (size_t)b * 32 * 3;
    float rx = rb[e * 3 + 0], ry = rb[e * 3 + 1], rz = rb[e * 3 + 2];
    const float* fr = fws + (size_t)pair * 39;

    float ps0 = 0.f, ps1 = 0.f, ps2 = 0.f;
    int row = 0;
    for (int j = 0; j < 32; ++j) {
        if (j == e) continue;
        float dx = rb[j * 3 + 0] - rx;
        float dy = rb[j * 3 + 1] - ry;
        float dz = rb[j * 3 + 2] - rz;
        float sq = dx * dx + dy * dy + dz * dz;
        float w = fr[row] / (1.f + sq * sqrtf(sq));
        ps0 += w * dx; ps1 += w * dy; ps2 += w * dz;
        ++row;
    }
    float decay = 1.f;
    for (int q = 0; q < 8; ++q) {
        float dx = rx - coords[q * 3 + 0];
        float dy = ry - coords[q * 3 + 1];
        float dz = rz - coords[q * 3 + 2];
        float sq = dx * dx + dy * dy + dz * dz;
        float w = fr[31 + q] / (1.f + sq * sqrtf(sq));
        ps0 += w * dx; ps1 += w * dy; ps2 += w * dz;
        float Z = chg[q];
        decay *= tanhf(sq * (4.f * Z * Z));         // sq / (0.5/Z)^2
    }
    float s = 0.030197383422318501f * decay;        // exp(-3.5)
    out[(size_t)pair * 3 + 0] = s * ps0;
    out[(size_t)pair * 3 + 1] = s * ps1;
    out[(size_t)pair * 3 + 2] = s * ps2;
}

// -------------------------------------------------------------------
extern "C" void kernel_launch(void* const* d_in, const int* in_sizes, int n_in,
                              void* d_out, int out_size, void* d_ws, size_t ws_size,
                              hipStream_t stream)
{
    const float* rs     = (const float*)d_in[0];
    const float* coords = (const float*)d_in[1];
    const float* mel    = (const float*)d_in[2];
    const float* mnuc   = (const float*)d_in[3];
    const float* emb    = (const float*)d_in[4];
    const float* w1el   = (const float*)d_in[5];
    const float* b1el   = (const float*)d_in[6];
    const float* w2el   = (const float*)d_in[7];
    const float* b2el   = (const float*)d_in[8];
    const float* w1nuc  = (const float*)d_in[9];
    const float* b1nuc  = (const float*)d_in[10];
    const float* w2nuc  = (const float*)d_in[11];
    const float* b2nuc  = (const float*)d_in[12];
    const float* chg    = (const float*)d_in[13];

    char* ws = (char*)d_ws;
    _Float16* wtel  = (_Float16*)ws;                       // 128 KB
    _Float16* wtnuc = wtel + 256 * 256;                    // 128 KB
    float*    fws   = (float*)(ws + 2 * 256 * 256 * sizeof(_Float16));

    prep_weights<<<256, 256, 0, stream>>>(w1el, w1nuc, wtel, wtnuc);

    const size_t lds_bytes = 3 * 256 * 128 * sizeof(_Float16);   // 192 KB
    // electron rows: 31 rows/pair, 8 pairs/WG -> 2048 WGs
    mlp_gemm<31, 8><<<N_PAIRS / 8, 512, lds_bytes, stream>>>(
        emb, mel, wtel, b1el, w2el, b2el, fws, 0);
    // nucleus rows: 8 rows/pair, 32 pairs/WG -> 512 WGs
    mlp_gemm<8, 32><<<N_PAIRS / 32, 512, lds_bytes, stream>>>(
        emb, mnuc, wtnuc, b1nuc, w2nuc, b2nuc, fws, 31);

    finalize<<<N_PAIRS / 256, 256, 0, stream>>>(rs, coords, chg, fws,
                                                (float*)d_out);
}
